// STGCN_61065845014838
// MI455X (gfx1250) — compile-verified, matrix operand order checked by reading, not run on hardware
//
#include <hip/hip_runtime.h>
#include <math.h>

typedef __attribute__((ext_vector_type(16))) _Float16 v16h;
typedef __attribute__((ext_vector_type(8)))  _Float16 v8h;
typedef __attribute__((ext_vector_type(2)))  __fp16   v2hf;   // cvt_pkrtz result type
typedef __attribute__((ext_vector_type(8)))  float    v8f;
typedef __attribute__((ext_vector_type(4)))  unsigned int u32x4;
typedef __attribute__((ext_vector_type(8)))  int          i32x8;
typedef __attribute__((ext_vector_type(4)))  int          i32x4;

#define TPB   256
#define BPW   16            // batches per workgroup
#define NV    16
#define NT    25
#define FEAT  800           // 32 channels * 25
#define GOUT  64
#define N2    128           // GOUT*2 (W0 | W1 fused)
#define HID   128
#define KFC   1024          // NV*GOUT
#define KCA   128           // cheb K chunk (4 wmma k-steps); tail chunk = 32
#define NCHA  7             // 6*128 + 32 = 800
#define KCFC  256           // fc K chunk (8 wmma k-steps)
#define HSTR  168           // padded f16 stride = 84 dwords (16B aligned, bank-clean)
#define ASTR  132           // padded f32 acc stride
#define GSTR  1032          // padded f16 stride for g
#define WFSTR 264           // padded f16 stride for fc weight chunk = 132 dwords

// ---- dynamic LDS layout (bytes) ----
#define OFF_X    0                       // 6400 f32 x tile
#define OFF_CW   (OFF_X   + 6400*4)
#define OFF_CB   (OFF_CW  + 96*4)
#define OFF_CHB  (OFF_CB  + 32*4)
#define OFF_ADJ  (OFF_CHB + 64*4)
#define OFF_DINV (OFF_ADJ + 256*4)
#define OFF_LH   (OFF_DINV+ 16*4)
#define OFF_G    (OFF_LH  + 272*4)       // 16*GSTR f16
#define OFF_LAT  (OFF_G   + 16*GSTR*2)   // 16*ASTR f32
#define OFF_U    (OFF_LAT + 16*ASTR*4)   // 70016
// phase A: hl [256][HSTR] f16 (86016B) then wl double buffer 2x[128][HSTR] f16 (2x43008B)
#define WLBUF    (128*HSTR*2)
#define WL_OFF(i) (OFF_U + 256*HSTR*2 + (i)*WLBUF)
// phase B: ac [256][ASTR] f32 (135168B) at OFF_U ; phase C: wf double buffer 2x[128][WFSTR] f16
#define WFBUF    (128*WFSTR*2)
#define WF_OFF(i) (OFF_U + (i)*WFBUF)
#define U_BYTES  (256*HSTR*2 + 2*WLBUF)  // 172032 >= 135168 (acc) >= 2*67584 (fc)
#define SMEM_BYTES (OFF_U + U_BYTES)     // 242048

// ---- workspace layout (prepacked f16, transposed weights) ----
#define WS_WT_ELTS (N2*FEAT)             // [128][800]
#define WS_WF_OFF  (WS_WT_ELTS)          // [128][1024] follows
#define WS_TOTAL   (WS_WT_ELTS + HID*KFC)

// TDM 2D tile load: rows x tile_k f16 elements, global row stride row_stride elts,
// LDS padded per D#.pad (pad_int_code: interval=2^(c+1) dwords, pad_amt_code: amount=c+1 dwords).
__device__ __forceinline__ void tdm_load_2d(unsigned lds_off, const _Float16* gsrc,
                                            unsigned tile_k, unsigned rows,
                                            unsigned row_stride,
                                            unsigned pad_int_code, unsigned pad_amt_code)
{
    unsigned long long ga = (unsigned long long)(uintptr_t)gsrc;
    u32x4 g0;
    g0[0] = 1u;                                   // count=1, load, user mode
    g0[1] = lds_off;                              // lds_addr (bytes)
    g0[2] = (unsigned)ga;                         // global_addr[31:0]
    g0[3] = (unsigned)((ga >> 32) & 0x01ffffffu) | 0x80000000u; // addr[56:32] | type=2
    i32x8 g1;
    g1[0] = (int)((1u << 16)                      // data_size = 2 bytes
                | (1u << 20)                      // pad_enable
                | (pad_int_code << 22)
                | (pad_amt_code << 25));
    g1[1] = (int)((tile_k & 0xffffu) << 16);                              // tensor_dim0 lo
    g1[2] = (int)(((tile_k >> 16) & 0xffffu) | ((rows & 0xffffu) << 16)); // dim0 hi | dim1 lo
    g1[3] = (int)(((rows >> 16) & 0xffffu) | ((tile_k & 0xffffu) << 16)); // dim1 hi | tile_dim0
    g1[4] = (int)(rows & 0xffffu);                                        // tile_dim1 (tile_dim2=0)
    g1[5] = (int)row_stride;                                              // dim0_stride lo32
    g1[6] = 0;
    g1[7] = 0;
    i32x4 z4 = {0, 0, 0, 0};
    i32x8 z8 = {0, 0, 0, 0, 0, 0, 0, 0};
    __builtin_amdgcn_tensor_load_to_lds(g0, g1, z4, z4, z8, 0);
}

__device__ __forceinline__ v16h frag16(const _Float16* p0, int stride, int lane, int col) {
    const _Float16* p = p0 + (lane & 15) * stride + col + ((lane >> 4) << 3);
    union { v16h v; v8h h[2]; } u;
    u.h[0] = *(const v8h*)(p);
    u.h[1] = *(const v8h*)(p + 16);
    return u.v;
}

__device__ __forceinline__ unsigned pack2h(float a, float b) {
    union { v2hf h; unsigned u; } cvt;
    cvt.h = __builtin_amdgcn_cvt_pkrtz(a, b);
    return cvt.u;
}

// one-time repack: (cheb_w0|cheb_w1)^T -> ws[0..] f16 [128][800]; fc_w^T -> f16 [128][1024]
__global__ void __launch_bounds__(TPB)
stgcn_prep(const float* __restrict__ cheb_w0, const float* __restrict__ cheb_w1,
           const float* __restrict__ fc_w, _Float16* __restrict__ ws)
{
    int idx = blockIdx.x * TPB + threadIdx.x;
    if (idx < WS_WT_ELTS) {
        int n = idx / FEAT, k = idx - n * FEAT;
        float v = (n < GOUT) ? cheb_w0[k*GOUT + n] : cheb_w1[k*GOUT + (n - GOUT)];
        ws[idx] = (_Float16)v;
    } else if (idx < WS_TOTAL) {
        int j = idx - WS_WT_ELTS;
        int n = j >> 10, k = j & (KFC - 1);
        ws[idx] = (_Float16)fc_w[k*HID + n];
    }
}

__global__ void __launch_bounds__(TPB)
stgcn_fused(const float* __restrict__ x, const int* __restrict__ eidx,
            const float* __restrict__ conv_w, const float* __restrict__ conv_b,
            const float* __restrict__ cheb_b,
            const float* __restrict__ fc_b,
            const float* __restrict__ final_w, const float* __restrict__ final_b,
            const _Float16* __restrict__ wt,   // [128][800] f16
            const _Float16* __restrict__ wfg,  // [128][1024] f16
            float* __restrict__ out)
{
    extern __shared__ char sm[];
    float*    xs  = (float*)(sm + OFF_X);
    float*    cw  = (float*)(sm + OFF_CW);
    float*    cb  = (float*)(sm + OFF_CB);
    float*    chb = (float*)(sm + OFF_CHB);
    float*    adj = (float*)(sm + OFF_ADJ);
    float*    dnv = (float*)(sm + OFF_DINV);
    float*    lh  = (float*)(sm + OFF_LH);
    _Float16* gl  = (_Float16*)(sm + OFF_G);
    float*    lat = (float*)(sm + OFF_LAT);
    _Float16* hl  = (_Float16*)(sm + OFF_U);   // phase A h chunk [256][HSTR]
    float*    ac  = (float*)(sm + OFF_U);      // phase B acc [256][ASTR]

    const int tid    = threadIdx.x;
    const int lane   = tid & 31;
    const int wave   = tid >> 5;
    const int batch0 = blockIdx.x * BPW;

    // prefetch first cheb weight chunk via TDM while phase 0 runs
    if (wave == 0)
        tdm_load_2d(WL_OFF(0), wt, KCA, N2, FEAT, 5u, 19u);   // 64dw rows + 20dw pad -> HSTR

    // ---------- phase 0: stage x tile + params, build Lhat ----------
    {
        const float* xsrc = x + (size_t)batch0 * NV * NT;
        #pragma unroll
        for (int i = 0; i < 25; ++i) xs[tid + TPB*i] = xsrc[tid + TPB*i];
        if (tid < 96) cw[tid]  = conv_w[tid];
        if (tid < 32) cb[tid]  = conv_b[tid];
        if (tid < 64) chb[tid] = cheb_b[tid];
        adj[tid] = 0.0f;
    }
    __syncthreads();
    if (tid < 64) {
        int s = eidx[tid] & 15;
        int d = eidx[64 + tid] & 15;
        atomicAdd(&adj[s*16 + d], 1.0f);
    }
    __syncthreads();
    if (tid < 16) {
        float dsum = 0.0f;
        for (int v = 0; v < 16; ++v) dsum += adj[tid*16 + v];
        dnv[tid] = dsum > 0.0f ? rsqrtf(dsum) : 0.0f;
    }
    __syncthreads();
    {
        int u = tid >> 4, v = tid & 15;
        lh[u*17 + v] = -dnv[u] * adj[u*16 + v] * dnv[v];
    }

    // ---------- phase A: fused conv + cheb GEMM [256,800] x [800,128] ----------
    const int mg = wave & 3;
    const int ng = wave >> 2;
    v8f acc[4][4];
    #pragma unroll
    for (int i = 0; i < 4; ++i)
        #pragma unroll
        for (int j = 0; j < 4; ++j)
            #pragma unroll
            for (int r = 0; r < 8; ++r) acc[i][j][r] = 0.0f;

    for (int chunk = 0; chunk < NCHA; ++chunk) {
        const int k0 = chunk * KCA;
        const int kc = (chunk < NCHA-1) ? KCA : 32;
        // conv1d (k=3, same pad) -> f16 h chunk, one row/thread, packed b32 stores
        {
            const float* xr = xs + tid * NT;
            unsigned*    hr = (unsigned*)(hl + tid * HSTR);
            int c = k0 / NT, t = k0 - (k0 / NT) * NT;
            for (int i = 0; i < kc/2; ++i) {
                float v0, v1;
                {
                    float xm = t ? xr[t-1] : 0.0f;
                    float x0 = xr[t];
                    float xp = (t < NT-1) ? xr[t+1] : 0.0f;
                    v0 = cb[c] + cw[3*c]*xm + cw[3*c+1]*x0 + cw[3*c+2]*xp;
                    if (++t == NT) { t = 0; ++c; }
                }
                {
                    float xm = t ? xr[t-1] : 0.0f;
                    float x0 = xr[t];
                    float xp = (t < NT-1) ? xr[t+1] : 0.0f;
                    v1 = cb[c] + cw[3*c]*xm + cw[3*c+1]*x0 + cw[3*c+2]*xp;
                    if (++t == NT) { t = 0; ++c; }
                }
                hr[i] = pack2h(v0, v1);
            }
        }
        // pipeline: issue next chunk's TDM, then wait for current chunk's DMA
        if (wave == 0) {
            if (chunk + 1 < NCHA) {
                const int nkc = (chunk + 1 < NCHA-1) ? KCA : 32;
                tdm_load_2d(WL_OFF((chunk+1) & 1), wt + (chunk+1)*KCA, nkc, N2, FEAT,
                            (nkc == KCA) ? 5u : 3u, (nkc == KCA) ? 19u : 67u);
                __builtin_amdgcn_s_wait_tensorcnt(1);
            } else {
                __builtin_amdgcn_s_wait_tensorcnt(0);
            }
        }
        __syncthreads();   // publish h chunk + TDM'd weight chunk

        const _Float16* wlp   = (const _Float16*)(sm + WL_OFF(chunk & 1));
        const _Float16* hbase = hl  + (mg*64)*HSTR;
        const _Float16* wbase = wlp + (ng*64)*HSTR;
        for (int ks = 0; ks < kc/32; ++ks) {
            const int col = ks*32;
            v16h bf[4];
            #pragma unroll
            for (int j = 0; j < 4; ++j) bf[j] = frag16(wbase + j*16*HSTR, HSTR, lane, col);
            #pragma unroll
            for (int i = 0; i < 4; ++i) {
                v16h a = frag16(hbase + i*16*HSTR, HSTR, lane, col);
                #pragma unroll
                for (int j = 0; j < 4; ++j)
                    acc[i][j] = __builtin_amdgcn_wmma_f32_16x16x32_f16(
                        false, a, false, bf[j], (short)0, acc[i][j], false, false);
            }
        }
        __syncthreads();
    }

    // ---------- phase B: spill acc, apply Lhat mix + bias -> g (f16) ----------
    {
        const int lr = lane >> 4, lc = lane & 15;
        #pragma unroll
        for (int i = 0; i < 4; ++i)
            #pragma unroll
            for (int j = 0; j < 4; ++j) {
                int m0 = mg*64 + i*16 + 8*lr;
                int n0 = ng*64 + j*16 + lc;
                #pragma unroll
                for (int r = 0; r < 8; ++r)
                    ac[(m0 + r)*ASTR + n0] = acc[i][j][r];
            }
    }
    __syncthreads();
    #pragma unroll
    for (int i = 0; i < (BPW*NV*GOUT/2)/TPB; ++i) {   // 32 iters, 2 adjacent n per iter
        int p  = tid + TPB*i;
        int n  = (p & 31) * 2;
        int u  = (p >> 5) & 15;
        int bl = p >> 9;
        const float* arow = ac + (bl*NV)*ASTR;
        float g0 = arow[u*ASTR + n]     + chb[n];
        float g1 = arow[u*ASTR + n + 1] + chb[n + 1];
        #pragma unroll
        for (int v = 0; v < 16; ++v) {
            float l = lh[u*17 + v];
            g0 += l * arow[v*ASTR + GOUT + n];
            g1 += l * arow[v*ASTR + GOUT + n + 1];
        }
        *(unsigned*)(gl + bl*GSTR + u*GOUT + n) = pack2h(g0, g1);
    }
    __syncthreads();

    // ---------- phase C: fc GEMM [16,1024] x [1024,128], TDM double-buffered ----------
    if (wave == 0)
        tdm_load_2d(WF_OFF(0), wfg, KCFC, HID, KFC, 6u, 3u);  // 128dw rows + 4dw pad -> WFSTR
    v8f afc;
    #pragma unroll
    for (int r = 0; r < 8; ++r) afc[r] = 0.0f;
    for (int cc = 0; cc < KFC/KCFC; ++cc) {
        if (wave == 0) {
            if (cc + 1 < KFC/KCFC) {
                tdm_load_2d(WF_OFF((cc+1) & 1), wfg + (cc+1)*KCFC, KCFC, HID, KFC, 6u, 3u);
                __builtin_amdgcn_s_wait_tensorcnt(1);
            } else {
                __builtin_amdgcn_s_wait_tensorcnt(0);
            }
        }
        __syncthreads();
        const _Float16* wfp = (const _Float16*)(sm + WF_OFF(cc & 1));
        for (int ks = 0; ks < KCFC/32; ++ks) {
            v16h a = frag16(gl, GSTR, lane, cc*KCFC + ks*32);
            v16h b = frag16(wfp + (wave*16)*WFSTR, WFSTR, lane, ks*32);
            afc = __builtin_amdgcn_wmma_f32_16x16x32_f16(
                false, a, false, b, (short)0, afc, false, false);
        }
        __syncthreads();
    }

    // ---------- phase D: relu + final dot + sigmoid ----------
    {
        const int lr = lane >> 4, lc = lane & 15;
        const int n0 = wave*16 + lc;
        #pragma unroll
        for (int r = 0; r < 8; ++r)
            lat[(r + 8*lr)*ASTR + n0] = afc[r];
    }
    __syncthreads();
    if (tid < BPW) {
        float s = final_b[0];
        for (int n = 0; n < HID; ++n) {
            float l = lat[tid*ASTR + n] + fc_b[n];
            l = l > 0.0f ? l : 0.0f;
            s += l * final_w[n];
        }
        out[batch0 + tid] = 1.0f / (1.0f + __expf(-s));
    }
}

extern "C" void kernel_launch(void* const* d_in, const int* in_sizes, int n_in,
                              void* d_out, int out_size, void* d_ws, size_t ws_size,
                              hipStream_t stream) {
    (void)in_sizes; (void)n_in; (void)ws_size; (void)out_size;
    _Float16* ws = (_Float16*)d_ws;
    stgcn_prep<<<dim3((WS_TOTAL + TPB - 1) / TPB), dim3(TPB), 0, stream>>>(
        (const float*)d_in[4],  // cheb_w0
        (const float*)d_in[5],  // cheb_w1
        (const float*)d_in[7],  // fc_w
        ws);
    stgcn_fused<<<dim3(8192 / BPW), dim3(TPB), SMEM_BYTES, stream>>>(
        (const float*)d_in[0],  // x
        (const int*)  d_in[1],  // edge_index
        (const float*)d_in[2],  // conv_w
        (const float*)d_in[3],  // conv_b
        (const float*)d_in[6],  // cheb_b
        (const float*)d_in[8],  // fc_b
        (const float*)d_in[9],  // final_w
        (const float*)d_in[10], // final_b
        ws,                     // cheb (W0|W1)^T f16
        ws + WS_WF_OFF,         // fc_w^T f16
        (float*)d_out);
}
